// LSTMModel_42992622632990
// MI455X (gfx1250) — matrix operand order repaired and checked by reading, not compiled, not run on hardware
//
#include <hip/hip_runtime.h>

// ---------------------------------------------------------------------------
// MI455X (gfx1250) LSTM:
//   phase 0: prep      - f32->bf16 copies of x / W_ih / W_hh, fused bias,
//                        zero h, reset barrier counter
//   phase 1: xproj     - x @ W_ih^T + bias  (bf16 WMMA, f32 accum) -> ws (f32)
//   phase 2: lstm_scan - persistent kernel, 512 steps, W_hh resident in LDS,
//                        c-state resident in VGPRs, async LDS fills,
//                        grid barrier per step
//   phase 3: final     - h_last @ W_lin^T + b_lin
// Roofline: 206 GFLOP total; phase 1 HBM-bound (~25us at 23.3 TB/s),
// phase 2 latency-bound (512 dependent steps) -> only h (128KB) moves/step.
// Workspace required: ~548 MB.
// ---------------------------------------------------------------------------

typedef __attribute__((ext_vector_type(16))) __bf16 v16bf;
typedef __attribute__((ext_vector_type(8)))  __bf16 v8bf;
typedef __attribute__((ext_vector_type(8)))  float  v8f;
typedef __attribute__((ext_vector_type(4)))  int    v4i_t;

#define INPUT_SZ  256
#define HIDDEN    512
#define GATES4    2048
#define BATCH     128
#define SEQ       512
#define NWG2      32          // workgroups in the persistent scan kernel

// ws layout (byte offsets, all 256-aligned)
#define XP_OFF    ((size_t)0)                          // f32 [SEQ][BATCH][GATES4] = 512MB
#define XP_BYTES  ((size_t)SEQ * BATCH * GATES4 * 4)
#define XBF_OFF   (XP_OFF + XP_BYTES)                  // bf16 [BATCH][SEQ][INPUT] = 32MB
#define XBF_BYTES ((size_t)BATCH * SEQ * INPUT_SZ * 2)
#define WIH_OFF   (XBF_OFF + XBF_BYTES)                // bf16 [2048][256] = 1MB
#define WIH_BYTES ((size_t)GATES4 * INPUT_SZ * 2)
#define WHH_OFF   (WIH_OFF + WIH_BYTES)                // bf16 [2048][512] = 2MB
#define WHH_BYTES ((size_t)GATES4 * HIDDEN * 2)
#define BIAS_OFF  (WHH_OFF + WHH_BYTES)                // f32 [2048]
#define BIAS_BYTES ((size_t)GATES4 * 4)
#define HG_OFF    (BIAS_OFF + BIAS_BYTES)              // bf16 [128][512]
#define HG_BYTES  ((size_t)BATCH * HIDDEN * 2)
#define HL_OFF    (HG_OFF + HG_BYTES)                  // f32 [128][512]
#define HL_BYTES  ((size_t)BATCH * HIDDEN * 4)
#define CNT_OFF   (HL_OFF + HL_BYTES)                  // u32 barrier counter

// LDS layout for the scan kernel: 16 B-panels of 16KB + 2 A-panels of 16KB
#define B_LDS_BYTES (16 * 16384)        // 256KB: W_hh^T, pre-swizzled
#define A_LDS_OFF   B_LDS_BYTES
#define SMEM_BYTES  (B_LDS_BYTES + 2 * 16384)   // 288KB <= 320KB/WGP

// CDNA5 async global->LDS copy (per-lane scatter, tracked by ASYNCcnt).
#if __has_builtin(__builtin_amdgcn_global_load_async_to_lds_b128) && \
    __has_builtin(__builtin_amdgcn_s_wait_asynccnt)
#define USE_ASYNC_LDS 1
#else
#define USE_ASYNC_LDS 0
#endif

static __device__ __forceinline__ void lds_fill16(const void* gsrc, void* ldst) {
#if USE_ASYNC_LDS
  __builtin_amdgcn_global_load_async_to_lds_b128(
      (__attribute__((address_space(1))) v4i_t*)(v4i_t*)gsrc,
      (__attribute__((address_space(3))) v4i_t*)(v4i_t*)ldst, 0, 0);
#else
  *(uint4*)ldst = *(const uint4*)gsrc;
#endif
}
static __device__ __forceinline__ void lds_fill_wait() {
#if USE_ASYNC_LDS
  __builtin_amdgcn_s_wait_asynccnt(0);
#endif
}

static __device__ __forceinline__ unsigned short f2bfb(float f) {
  // round-to-nearest-even f32 -> bf16 bits (cold paths only)
  unsigned u = __builtin_bit_cast(unsigned, f);
  u += 0x7FFFu + ((u >> 16) & 1u);
  return (unsigned short)(u >> 16);
}
static __device__ __forceinline__ float sigf(float x) {
  return 1.0f / (1.0f + __expf(-x));
}
static __device__ __forceinline__ float tanh_fast(float x) {
  // 1 - 2/(e^{2x}+1): saturates correctly at +-1 for large |x|
  return 1.0f - 2.0f / (__expf(2.0f * x) + 1.0f);
}

// ---------------------------------------------------------------------------
// phase 0: prep (one-shot conversions; bandwidth-trivial)
// ---------------------------------------------------------------------------
__global__ void prep_kernel(const float* __restrict__ x, const float* __restrict__ Wih,
                            const float* __restrict__ Whh,
                            const float* __restrict__ bih, const float* __restrict__ bhh,
                            unsigned short* __restrict__ xbf,
                            unsigned short* __restrict__ wih_bf, unsigned short* __restrict__ whh_bf,
                            float* __restrict__ bias, unsigned short* __restrict__ hglob,
                            unsigned* __restrict__ cnt) {
  const int gid = blockIdx.x * blockDim.x + threadIdx.x;
  const int stride = gridDim.x * blockDim.x;
  // x: 16.7M elements, vectorized 4-wide
  const float4*  x4 = (const float4*)x;
  ushort4* o4 = (ushort4*)xbf;
  for (int i = gid; i < (BATCH * SEQ * INPUT_SZ) / 4; i += stride) {
    const float4 v = x4[i];
    ushort4 o;
    o.x = f2bfb(v.x); o.y = f2bfb(v.y); o.z = f2bfb(v.z); o.w = f2bfb(v.w);
    o4[i] = o;
  }
  for (int i = gid; i < GATES4 * INPUT_SZ; i += stride) wih_bf[i] = f2bfb(Wih[i]);
  for (int i = gid; i < GATES4 * HIDDEN;   i += stride) whh_bf[i] = f2bfb(Whh[i]);
  for (int i = gid; i < GATES4;            i += stride) bias[i] = bih[i] + bhh[i];
  for (int i = gid; i < BATCH * HIDDEN;    i += stride) hglob[i] = 0;   // h0 = 0
  if (gid == 0) *cnt = 0;                                               // barrier reset
}

// ---------------------------------------------------------------------------
// phase 1: x_proj[t][b][n] = sum_k x[b][t][k] * W_ih[n][k] + bias[n]
// One wave -> 16(M) x 64(N) strip, K=256 in 8 chunks of 32.
// A: bf16 x, two 16B loads per lane in contiguous k order (k = kt*32 + 16*hi + s).
// B: W_ih row-major bf16 gives B in the same contiguous k order (32 contiguous bytes/lane).
// ---------------------------------------------------------------------------
__global__ void __launch_bounds__(256)
xproj_kernel(const unsigned short* __restrict__ xbf, const unsigned short* __restrict__ wih_bf,
             const float* __restrict__ bias, float* __restrict__ xp) {
  const int wid  = (blockIdx.x * blockDim.x + threadIdx.x) >> 5;  // 131072 waves
  const int lane = threadIdx.x & 31;
  const int mt = wid >> 5;          // 0..4095 : 16 rows of [B*S, 256]
  const int ns = wid & 31;          // n-strip : n0 = ns*64
  const int col = lane & 15, hi = lane >> 4;

  v8f acc[4] = {};
  const int m = mt * 16 + col;                       // A row for this lane
  const unsigned short* xrow = xbf + (size_t)m * INPUT_SZ;

#pragma unroll
  for (int kt = 0; kt < 8; ++kt) {
    // A tile: half h -> k = kt*32 + h + 16*hi
    v8bf alo = *(const v8bf*)(xrow + kt * 32 + 16 * hi);
    v8bf ahi = *(const v8bf*)(xrow + kt * 32 + 16 * hi + 8);
    v16bf a = __builtin_shufflevector(alo, ahi, 0, 1, 2, 3, 4, 5, 6, 7,
                                      8, 9, 10, 11, 12, 13, 14, 15);
#pragma unroll
    for (int nt = 0; nt < 4; ++nt) {
      const int n = ns * 64 + nt * 16 + col;         // B column (= W_ih row)
      v16bf b = *(const v16bf*)(wih_bf + (size_t)n * INPUT_SZ + kt * 32 + 16 * hi);
      acc[nt] = __builtin_amdgcn_wmma_f32_16x16x32_bf16(false, a, false, b,
                                                        (short)0, acc[nt], false, false);
    }
  }
  // bias + scatter-store into [t][b][4H] layout for the scan kernel
#pragma unroll
  for (int nt = 0; nt < 4; ++nt) {
    const int n = ns * 64 + nt * 16 + col;
    const float bv = bias[n];
#pragma unroll
    for (int r = 0; r < 8; ++r) {
      const int mrow = mt * 16 + r + 8 * hi;         // global row = b*SEQ + s
      const int b_ = mrow >> 9, s_ = mrow & (SEQ - 1);
      xp[((size_t)(s_ * BATCH + b_)) * GATES4 + n] = acc[nt][r] + bv;
    }
  }
}

// ---------------------------------------------------------------------------
// phase 2: persistent recurrent scan.
// 32 WGs x 256 thr (8 waves). WG = 2 m-tiles x 4 j-tiles; wave = 16x16 (b,j)
// h/c tile, computes i,f,g,o gate tiles for it. W_hh^T panels pre-swizzled
// into LDS once (async); h tile re-swizzled into LDS each step (async).
// Grid barrier per step; next-step x_proj prefetched under the barrier.
// ---------------------------------------------------------------------------
__global__ void __launch_bounds__(256)
lstm_scan_kernel(const float* __restrict__ xp, const unsigned short* __restrict__ whh_bf,
                 unsigned short* __restrict__ hglob, float* __restrict__ hlast,
                 unsigned* __restrict__ cnt) {
  extern __shared__ char smem[];     // [0,256K): B panels ; [256K,288K): A panels
  const int tid = threadIdx.x;
  const int lane = tid & 31, wv = tid >> 5;
  const int mgrp = blockIdx.x >> 3, jgrp = blockIdx.x & 7;
  const int mbase = mgrp * 32, jbase = jgrp * 64;
  const int mtl = wv >> 2, jtl = wv & 3;
  const int m0 = mbase + mtl * 16, j0 = jbase + jtl * 16;
  const int col = lane & 15, hi = lane >> 4;

  // One-time fill: 16 B-panels (panel p = jtl*4 + gate), swizzled so a lane's
  // v16bf is 32 contiguous bytes: half h -> k = kt*32 + h + 16*(lane>>4).
  for (int c = tid; c < 16384; c += 256) {            // 64 x 16B chunks/thread
    const int p = c >> 10, r = c & 1023, jc = r >> 6, k8 = r & 63;
    const int g = p & 3, jt = p >> 2;
    const int n = g * HIDDEN + jbase + jt * 16 + jc;  // W_hh row
    const int kt = k8 >> 2, rem = k8 & 3, bhi = rem >> 1, hseg = rem & 1;
    lds_fill16(whh_bf + (size_t)n * HIDDEN + k8 * 8,
               smem + p * 16384 + kt * 1024 + (jc + 16 * bhi) * 32 + hseg * 16);
  }

  v8f cst = {};                                       // cell state, VGPR-resident
  for (int t = 0; t < SEQ; ++t) {
    __syncthreads();                                  // A-LDS free to overwrite
    // Fill A panels from h (rows mbase..mbase+31), swizzled:
    // half h -> k = kt*32 + h + 16*(lane>>4)
    for (int c = tid; c < 2048; c += 256) {           // 8 x 16B chunks/thread
      const int ml = c >> 6, k8 = c & 63;
      const int kt = k8 >> 2, rem = k8 & 3, ahi = rem >> 1, hseg = rem & 1;
      lds_fill16(hglob + (size_t)(mbase + ml) * HIDDEN + k8 * 8,
                 smem + A_LDS_OFF + (ml >> 4) * 16384 + kt * 1024 +
                     ((ml & 15) + 16 * ahi) * 32 + hseg * 16);
    }
    lds_fill_wait();                                  // ASYNCcnt -> 0 (covers B at t=0)
    __syncthreads();

    // C-init from x_proj (coalesced 64B segments across lanes)
    const float* xpt = xp + (size_t)t * BATCH * GATES4;
    v8f acc[4];
#pragma unroll
    for (int g = 0; g < 4; ++g)
#pragma unroll
      for (int r = 0; r < 8; ++r)
        acc[g][r] = xpt[(size_t)(m0 + r + 8 * hi) * GATES4 + g * HIDDEN + j0 + col];

    // K loop: 16 x (1 A ds-load + 4 x (B ds-load + WMMA))
    const char* Ab = smem + A_LDS_OFF + mtl * 16384;
    const char* Bb = smem + (jtl * 4) * 16384;
#pragma unroll 4
    for (int kt = 0; kt < 16; ++kt) {
      v16bf a = *(const v16bf*)(Ab + kt * 1024 + lane * 32);
#pragma unroll
      for (int g = 0; g < 4; ++g) {
        v16bf b = *(const v16bf*)(Bb + g * 16384 + kt * 1024 + lane * 32);
        acc[g] = __builtin_amdgcn_wmma_f32_16x16x32_bf16(false, a, false, b,
                                                         (short)0, acc[g], false, false);
      }
    }

    // Elementwise gate mix; c stays in registers, h goes out as bf16
#pragma unroll
    for (int r = 0; r < 8; ++r) {
      const float gi = sigf(acc[0][r]);
      const float gf = sigf(acc[1][r]);
      const float gg = tanh_fast(acc[2][r]);
      const float go = sigf(acc[3][r]);
      const float cc = gf * cst[r] + gi * gg;
      cst[r] = cc;
      const float hv = go * tanh_fast(cc);
      const size_t hidx = (size_t)(m0 + r + 8 * hi) * HIDDEN + j0 + col;
      hglob[hidx] = f2bfb(hv);
      if (t == SEQ - 1) hlast[hidx] = hv;             // full-precision for phase 3
    }

    // Prefetch next step's x_proj tiles into cache while we sit in the barrier
    if (t + 1 < SEQ) {
      const float* xpn = xp + (size_t)(t + 1) * BATCH * GATES4;
#pragma unroll
      for (int g = 0; g < 4; ++g)
#pragma unroll
        for (int r = 0; r < 8; r += 2)
          __builtin_prefetch(&xpn[(size_t)(m0 + r + 8 * hi) * GATES4 + g * HIDDEN + j0 + col], 0, 3);
    }

    // Grid-wide release/acquire barrier (monotone counter; reset by prep)
    __threadfence();
    __syncthreads();
    if (tid == 0) {
      __hip_atomic_fetch_add(cnt, 1u, __ATOMIC_RELEASE, __HIP_MEMORY_SCOPE_AGENT);
      const unsigned target = (unsigned)(NWG2 * (t + 1));
      while (__hip_atomic_load(cnt, __ATOMIC_ACQUIRE, __HIP_MEMORY_SCOPE_AGENT) < target)
        __builtin_amdgcn_s_sleep(1);
    }
    __syncthreads();
  }
}

// ---------------------------------------------------------------------------
// phase 3: out[b] = h_last[b,:] . W_lin + b_lin   (128 x 512, trivial)
// ---------------------------------------------------------------------------
__global__ void final_linear_kernel(const float* __restrict__ hlast,
                                    const float* __restrict__ Wlin,
                                    const float* __restrict__ blin,
                                    float* __restrict__ out) {
  const int b = threadIdx.x;                          // 128 threads
  const float* hr = hlast + (size_t)b * HIDDEN;
  float s = 0.f;
  for (int j = 0; j < HIDDEN; ++j) s += hr[j] * Wlin[j];
  out[b] = s + blin[0];
}

// ---------------------------------------------------------------------------
extern "C" void kernel_launch(void* const* d_in, const int* in_sizes, int n_in,
                              void* d_out, int out_size, void* d_ws, size_t ws_size,
                              hipStream_t stream) {
  (void)in_sizes; (void)n_in; (void)out_size; (void)ws_size;
  const float* x    = (const float*)d_in[0];
  const float* Wih  = (const float*)d_in[1];
  const float* Whh  = (const float*)d_in[2];
  const float* bih  = (const float*)d_in[3];
  const float* bhh  = (const float*)d_in[4];
  const float* Wlin = (const float*)d_in[5];
  const float* blin = (const float*)d_in[6];
  float* out = (float*)d_out;

  char* ws = (char*)d_ws;
  float*          xp     = (float*)(ws + XP_OFF);
  unsigned short* xbf    = (unsigned short*)(ws + XBF_OFF);
  unsigned short* wih_bf = (unsigned short*)(ws + WIH_OFF);
  unsigned short* whh_bf = (unsigned short*)(ws + WHH_OFF);
  float*          bias   = (float*)(ws + BIAS_OFF);
  unsigned short* hglob  = (unsigned short*)(ws + HG_OFF);
  float*          hlast  = (float*)(ws + HL_OFF);
  unsigned*       cnt    = (unsigned*)(ws + CNT_OFF);

  prep_kernel<<<512, 256, 0, stream>>>(x, Wih, Whh, bih, bhh, xbf, wih_bf, whh_bf,
                                       bias, hglob, cnt);
  // 4096 m-tiles x 32 n-strips = 131072 waves -> 16384 blocks of 8 waves
  xproj_kernel<<<16384, 256, 0, stream>>>(xbf, wih_bf, bias, xp);
  lstm_scan_kernel<<<NWG2, 256, SMEM_BYTES, stream>>>(xp, whh_bf, hglob, hlast, cnt);
  final_linear_kernel<<<1, 128, 0, stream>>>(hlast, Wlin, blin, out);
}
